// SVHNQ_19430432047479
// MI455X (gfx1250) — 8-bit matrix operand order repaired and checked by reading, compiled, not run on hardware
//
#include <hip/hip_runtime.h>
#include <hip/hip_bf16.h>
#include <stdint.h>


typedef __attribute__((ext_vector_type(8))) int v8i;
typedef __attribute__((ext_vector_type(2))) int v2i;

#define BN_EPS 1e-5f

__host__ __device__ constexpr int a_off_c(int kk, int Ci, int Kw, int Wp) {
  int q  = kk / Ci;
  int ci = kk - q * Ci;
  int kh = q / Kw;
  int kw = q - kh * Kw;
  return (kh * Wp + kw) * Ci + ci;
}

__global__ void fill_zero_kernel(uint32_t* __restrict__ p, int ndw) {
  int i = blockIdx.x * blockDim.x + threadIdx.x;
  if (i < ndw) p[i] = 0;
}

__global__ void absmax_tanh_kernel(const float* __restrict__ w, int n,
                                   float* __restrict__ out) {
  __shared__ float red[256];
  float m = 0.f;
  for (int i = threadIdx.x; i < n; i += 256) m = fmaxf(m, fabsf(tanhf(w[i])));
  red[threadIdx.x] = m;
  __syncthreads();
  for (int s = 128; s > 0; s >>= 1) {
    if (threadIdx.x < s)
      red[threadIdx.x] = fmaxf(red[threadIdx.x], red[threadIdx.x + s]);
    __syncthreads();
  }
  if (threadIdx.x == 0) out[0] = red[0];
}

__global__ void quantw_kernel(const float* __restrict__ w,
                              const float* __restrict__ amax,
                              int8_t* __restrict__ qw,
                              int Ci, int KHW, int K64, int total) {
  int i = blockIdx.x * blockDim.x + threadIdx.x;
  if (i >= total) return;
  int K  = Ci * KHW;
  int co = i / K64;
  int r  = i - co * K64;
  int8_t val = 0;
  if (r < K) {
    int khkw = r / Ci;
    int ci   = r - khkw * Ci;
    float t = tanhf(w[((size_t)co * Ci + ci) * KHW + khkw]);
    float u = t / (2.f * amax[0]) + 0.5f;
    float q = rintf(u * 15.f);
    q = fminf(fmaxf(q, 0.f), 15.f);
    val = (int8_t)(2 * (int)q - 15);
  }
  qw[i] = val;
}

__global__ void conv1_pool_act_kernel(const float* __restrict__ x,
                                      const float* __restrict__ W0,
                                      const float* __restrict__ b0,
                                      uint8_t* __restrict__ act) {
  __shared__ float sw[24 * 75];
  for (int i = threadIdx.x; i < 24 * 75; i += blockDim.x) sw[i] = W0[i];
  __syncthreads();
  int idx = blockIdx.x * blockDim.x + threadIdx.x;
  const int total = 1024 * 18 * 18 * 24;
  if (idx >= total) return;
  int c  = idx % 24; int t = idx / 24;
  int pw = t % 18;   t /= 18;
  int ph = t % 18;   int n = t / 18;
  float bias = b0[c];
  float s00 = bias, s01 = bias, s10 = bias, s11 = bias;
  const float* wc = &sw[c * 75];
  const float* xb = x + (size_t)n * 3 * 40 * 40;
  const int oh = 2 * ph, ow = 2 * pw;
  for (int ci = 0; ci < 3; ++ci) {
    const float* xc = xb + ci * 1600;
    const float* wk = wc + ci * 25;
#pragma unroll
    for (int kh = 0; kh < 5; ++kh) {
      const float* xr0 = xc + (oh + kh) * 40 + ow;
      const float* xr1 = xr0 + 40;
      const float* wr  = wk + kh * 5;
#pragma unroll
      for (int kw = 0; kw < 5; ++kw) {
        float wv = wr[kw];
        s00 = fmaf(xr0[kw],     wv, s00);
        s01 = fmaf(xr0[kw + 1], wv, s01);
        s10 = fmaf(xr1[kw],     wv, s10);
        s11 = fmaf(xr1[kw + 1], wv, s11);
      }
    }
  }
  float m = fmaxf(fmaxf(s00, s01), fmaxf(s10, s11));
  m = fminf(fmaxf(m, 0.f), 1.f);
  act[(((size_t)n * 20 + ph + 1) * 20 + pw + 1) * 24 + c] =
      (uint8_t)(int)rintf(m * 15.f);
}

template <int NB, int Ci, int Hi, int Wi, int Co, int Kh, int Kw, int PAD,
          int Ho, int Wo>
__global__ void convq_wmma_kernel(const uint8_t* __restrict__ act,
                                  const int8_t* __restrict__ qw,
                                  float* __restrict__ out) {
  constexpr int K    = Kh * Kw * Ci;
  constexpr int NCH  = (K + 63) / 64;
  constexpr int K64  = NCH * 64;
  constexpr int HoWo = Ho * Wo;
  constexpr int Hp   = Hi + 2 * PAD;
  constexpr int Wp   = Wi + 2 * PAD;
  __shared__ alignas(16) uint8_t sB[16 * K64];

  const int tid  = threadIdx.x;
  const int lane = tid & 31;
  const int wave = tid >> 5;
  const int mt0  = (blockIdx.x * 8 + wave) * 4;
  const int cbase = blockIdx.y * 16;

  {
    const int* src = (const int*)(qw + (size_t)cbase * K64);
    int* dst = (int*)sB;
    for (int d = tid; d < 16 * K64 / 4; d += 256) dst[d] = src[d];
  }
  __syncthreads();

  const int row  = lane & 15;
  const int half = lane >> 4;

  int ab[4];
#pragma unroll
  for (int t = 0; t < 4; ++t) {
    int p  = (mt0 + t) * 16 + row;
    int ni = p / HoWo;
    int pr = p - ni * HoWo;
    int oh = pr / Wo, ow = pr - oh * Wo;
    ab[t] = ((ni * Hp + oh) * Wp + ow) * Ci;
  }
  const uint8_t* brow = sB + row * K64 + (half << 4);

  v8i acc[4] = {};
#pragma unroll
  for (int ck = 0; ck < NCH; ++ck) {
    v8i b;
#pragma unroll
    for (int i = 0; i < 8; ++i) {
      int kb = ck * 64 + ((i >> 2) << 5) + ((i & 3) << 2);
      b[i] = *(const int*)(brow + kb);
    }
    v8i a[4];
#pragma unroll
    for (int j = 0; j < 4; ++j) {
      int kk0 = ck * 64 + ((j >> 1) << 5) + ((j & 1) << 3);
      int o0 = (kk0     < K) ? a_off_c(kk0,     Ci, Kw, Wp) : 0;
      int o1 = (kk0 + 16 < K) ? a_off_c(kk0 + 16, Ci, Kw, Wp) : 0;
      int off = half ? o1 : o0;
#pragma unroll
      for (int t = 0; t < 4; ++t) {
        v2i v = *(const v2i*)(act + (ab[t] + off));
        a[t][2 * j]     = v.x;
        a[t][2 * j + 1] = v.y;
      }
    }
#pragma unroll
    for (int t = 0; t < 4; ++t)
      acc[t] = __builtin_amdgcn_wmma_i32_16x16x64_iu8(false, a[t], true, b,
                                                      acc[t], false, false);
  }

  const float inv225 = 1.f / 225.f;
#pragma unroll
  for (int t = 0; t < 4; ++t) {
#pragma unroll
    for (int r = 0; r < 8; ++r) {
      int pp = (mt0 + t) * 16 + r + (half << 3);
      out[(size_t)pp * Co + cbase + row] = (float)acc[t][r] * inv225;
    }
  }
}

__global__ void bnstats_kernel(const float* __restrict__ x, int NHW, int C,
                               float* __restrict__ mean,
                               float* __restrict__ var) {
  __shared__ float rs[256], rs2[256];
  int c = blockIdx.x;
  float s = 0.f, s2 = 0.f;
  for (int t = threadIdx.x; t < NHW; t += 256) {
    float v = x[(size_t)t * C + c];
    s += v; s2 += v * v;
  }
  rs[threadIdx.x] = s; rs2[threadIdx.x] = s2;
  __syncthreads();
  for (int st = 128; st > 0; st >>= 1) {
    if (threadIdx.x < st) {
      rs[threadIdx.x]  += rs[threadIdx.x + st];
      rs2[threadIdx.x] += rs2[threadIdx.x + st];
    }
    __syncthreads();
  }
  if (threadIdx.x == 0) {
    float m = rs[0] / (float)NHW;
    mean[c] = m;
    var[c]  = fmaxf(rs2[0] / (float)NHW - m * m, 0.f);
  }
}

__global__ void bn_act_kernel(const float* __restrict__ x,
                              const float* __restrict__ mean,
                              const float* __restrict__ var,
                              const float* __restrict__ g,
                              const float* __restrict__ be,
                              uint8_t* __restrict__ act,
                              int N, int C, int H, int W, int pool, int halo) {
  int Ho = pool ? H / 2 : H, Wo = pool ? W / 2 : W;
  int total = N * Ho * Wo * C;
  int idx = blockIdx.x * blockDim.x + threadIdx.x;
  if (idx >= total) return;
  int c = idx % C; int t = idx / C;
  int w = t % Wo;  t /= Wo;
  int h = t % Ho;  int n = t / Ho;
  float rstd = rsqrtf(var[c] + BN_EPS);
  float sc = g[c] * rstd;
  float sh = be[c] - mean[c] * sc;
  const float* xb = x + ((size_t)n * H * W) * C + c;
  float v;
  if (pool) {
    float a = xb[((2*h)   * W + 2*w    ) * C] * sc + sh;
    float b = xb[((2*h)   * W + 2*w + 1) * C] * sc + sh;
    float d = xb[((2*h+1) * W + 2*w    ) * C] * sc + sh;
    float e = xb[((2*h+1) * W + 2*w + 1) * C] * sc + sh;
    v = fmaxf(fmaxf(a, b), fmaxf(d, e));
  } else {
    v = xb[(h * W + w) * C] * sc + sh;
  }
  v = fminf(fmaxf(v, 0.f), 1.f);
  int Wp = Wo + 2 * halo;
  act[(((size_t)n * (Ho + 2 * halo) + h + halo) * Wp + w + halo) * C + c] =
      (uint8_t)(int)rintf(v * 15.f);
}

__global__ void fc_kernel(const float* __restrict__ h,
                          const float* __restrict__ Wfc,
                          const float* __restrict__ bfc,
                          float* __restrict__ out) {
  int idx = blockIdx.x * blockDim.x + threadIdx.x;
  if (idx >= 1024 * 10) return;
  int j = idx % 10, n = idx / 10;
  const float* hp = h + (size_t)n * 256;
  const float* wp = Wfc + (size_t)j * 256;
  float s = bfc[j];
#pragma unroll 8
  for (int k = 0; k < 256; ++k) {
    float v = fminf(fmaxf(hp[k], 0.f), 1.f);
    s = fmaf(v, wp[k], s);
  }
  out[idx] = s;
}

template <int NB, int Ci, int Hi, int Wi, int Co, int Kh, int Kw, int PAD,
          int Ho, int Wo>
static void launch_convq(const uint8_t* a, const int8_t* w, float* o,
                         hipStream_t s) {
  constexpr int Mtot = NB * Ho * Wo;
  dim3 grid(Mtot / 512, Co / 16);
  convq_wmma_kernel<NB, Ci, Hi, Wi, Co, Kh, Kw, PAD, Ho, Wo>
      <<<grid, 256, 0, s>>>(a, w, o);
}

static void run_bn_act(const float* conv, int N, int C, int H, int W,
                       const float* g, const float* be, float* mean,
                       float* var, uint8_t* actout, int pool, int halo,
                       hipStream_t stream) {
  bnstats_kernel<<<C, 256, 0, stream>>>(conv, N * H * W, C, mean, var);
  int total = pool ? N * (H / 2) * (W / 2) * C : N * H * W * C;
  bn_act_kernel<<<(total + 255) / 256, 256, 0, stream>>>(
      conv, mean, var, g, be, actout, N, C, H, W, pool, halo);
}

static void fill_zero(void* p, size_t bytes, hipStream_t stream) {
  int ndw = (int)(bytes / 4);
  fill_zero_kernel<<<(ndw + 255) / 256, 256, 0, stream>>>((uint32_t*)p, ndw);
}

extern "C" void kernel_launch(void* const* d_in, const int* in_sizes, int n_in,
                              void* d_out, int out_size, void* d_ws,
                              size_t ws_size, hipStream_t stream) {
  const float* x   = (const float*)d_in[0];
  const float* W0  = (const float*)d_in[1];
  const float* b0  = (const float*)d_in[2];
  const float* Wfc = (const float*)d_in[19];
  const float* bfc = (const float*)d_in[20];

  uint8_t* ws = (uint8_t*)d_ws;
  int8_t* qw1 = (int8_t*)(ws + 0);
  int8_t* qw2 = (int8_t*)(ws + 8192);
  int8_t* qw3 = (int8_t*)(ws + 20480);
  int8_t* qw4 = (int8_t*)(ws + 40960);
  int8_t* qw5 = (int8_t*)(ws + 81920);
  int8_t* qw6 = (int8_t*)(ws + 122880);
  float* amax = (float*)(ws + 540672);
  float* mean = (float*)(ws + 541696);
  float* var  = (float*)(ws + 542720);
  uint8_t* actA = ws + (1u << 20);
  uint8_t* actB = ws + (1u << 20) + 16u * 1024 * 1024;
  float*   conv = (float*)(ws + (1u << 20) + 32u * 1024 * 1024);

  const int widx[6] = {3, 6, 9, 12, 15, 18};
  const int wci[6]  = {24, 32, 32, 64, 64, 64};
  const int wkk[6]  = {9, 9, 9, 9, 9, 25};
  const int wco[6]  = {32, 32, 64, 64, 64, 256};
  const int wk64[6] = {256, 320, 320, 576, 576, 1600};
  int8_t* qwp[6]    = {qw1, qw2, qw3, qw4, qw5, qw6};
  for (int i = 0; i < 6; ++i) {
    const float* w = (const float*)d_in[widx[i]];
    int nsrc = wco[i] * wci[i] * wkk[i];
    int ndst = wco[i] * wk64[i];
    absmax_tanh_kernel<<<1, 256, 0, stream>>>(w, nsrc, amax + i);
    quantw_kernel<<<(ndst + 255) / 256, 256, 0, stream>>>(
        w, amax + i, qwp[i], wci[i], wkk[i], wk64[i], ndst);
  }

  fill_zero(actA, (size_t)1024 * 20 * 20 * 24, stream);
  {
    int total = 1024 * 18 * 18 * 24;
    conv1_pool_act_kernel<<<(total + 255) / 256, 256, 0, stream>>>(x, W0, b0,
                                                                   actA);
  }

  launch_convq<1024, 24, 18, 18, 32, 3, 3, 1, 18, 18>(actA, qw1, conv, stream);
  fill_zero(actB, (size_t)1024 * 20 * 20 * 32, stream);
  run_bn_act(conv, 1024, 32, 18, 18, (const float*)d_in[4],
             (const float*)d_in[5], mean, var, actB, 0, 1, stream);
  launch_convq<1024, 32, 18, 18, 32, 3, 3, 1, 18, 18>(actB, qw2, conv, stream);
  run_bn_act(conv, 1024, 32, 18, 18, (const float*)d_in[7],
             (const float*)d_in[8], mean, var, actA, 1, 0, stream);
  launch_convq<1024, 32, 9, 9, 64, 3, 3, 0, 7, 7>(actA, qw3, conv, stream);
  fill_zero(actB, (size_t)1024 * 9 * 9 * 64, stream);
  run_bn_act(conv, 1024, 64, 7, 7, (const float*)d_in[10],
             (const float*)d_in[11], mean, var, actB, 0, 1, stream);
  launch_convq<1024, 64, 7, 7, 64, 3, 3, 1, 7, 7>(actB, qw4, conv, stream);
  run_bn_act(conv, 1024, 64, 7, 7, (const float*)d_in[13],
             (const float*)d_in[14], mean, var, actA, 0, 0, stream);
  launch_convq<1024, 64, 7, 7, 64, 3, 3, 0, 5, 5>(actA, qw5, conv, stream);
  run_bn_act(conv, 1024, 64, 5, 5, (const float*)d_in[16],
             (const float*)d_in[17], mean, var, actB, 0, 0, stream);
  launch_convq<1024, 64, 5, 5, 256, 5, 5, 0, 1, 1>(actB, qw6, conv, stream);

  fc_kernel<<<(1024 * 10 + 255) / 256, 256, 0, stream>>>(conv, Wfc, bfc,
                                                         (float*)d_out);
}
